// MultiHeadAttention_28381143892169
// MI455X (gfx1250) — matrix operand order repaired and checked by reading, not compiled, not run on hardware
//
#include <hip/hip_runtime.h>

typedef __bf16 bf16;
typedef __attribute__((ext_vector_type(16))) __bf16 v16bf;
typedef __attribute__((ext_vector_type(8)))  __bf16 v8bf;
typedef __attribute__((ext_vector_type(2)))  __bf16 bf16x2;
typedef __attribute__((ext_vector_type(8)))  float  v8f;
typedef __attribute__((ext_vector_type(8)))  int    v8i;

#define D_MODEL 1024
#define NHEAD   16
#define DH      64
#define SEQ     2048
#define BATCH   4
#define MTOT    (BATCH * SEQ)   // 8192

// ---------------------------------------------------------------------------
// f32 -> bf16 conversion (grid-stride)
// ---------------------------------------------------------------------------
__global__ void __launch_bounds__(256)
cvt_f32_bf16(const float* __restrict__ src, bf16* __restrict__ dst, int n) {
    int i = blockIdx.x * blockDim.x + threadIdx.x;
    int stride = gridDim.x * blockDim.x;
    for (; i < n; i += stride) dst[i] = (bf16)src[i];
}

// ---------------------------------------------------------------------------
// GEMM: Y[m,n] = (sum_k A[m,k] * W[n,k] + bias[n]) * scale
//   A: (MTOT, D_MODEL) bf16 row-major, W: (D_MODEL, D_MODEL) bf16 row-major.
// Block = 256 threads = 8 waves; wave computes 16(M) x 64(N); K step = 32.
// MODE 0: write bf16 to (B,H,S,Dh)      (Q pre-scaled, K)
// MODE 1: write bf16 to (B,H,Dh,S)      (V transposed)
// MODE 2: write f32  to (MTOT, D_MODEL) (final output)
// ---------------------------------------------------------------------------
template <int MODE>
__global__ void __launch_bounds__(256)
gemm_bias_wmma(const bf16* __restrict__ A, const bf16* __restrict__ W,
               const float* __restrict__ bias, void* __restrict__ out,
               float scale) {
    const int lane = threadIdx.x & 31;
    const int wave = threadIdx.x >> 5;
    const int half = lane >> 4;      // 0: lanes 0-15, 1: lanes 16-31
    const int ln   = lane & 15;
    const int m0   = blockIdx.x * 128 + wave * 16;
    const int n0   = blockIdx.y * 64;

    v8f acc[4] = {};

    const bf16* aptr = A + (size_t)(m0 + ln) * D_MODEL;

    for (int kk = 0; kk < D_MODEL; kk += 32) {
        // A fragment (16x32 bf16): lanes 0-15 K=kk..kk+15,
        //                          lanes 16-31 K=kk+16..kk+31 (contiguous)
        v8bf alo = *(const v8bf*)(aptr + kk + (half ? 16 : 0));
        v8bf ahi = *(const v8bf*)(aptr + kk + 8 + (half ? 16 : 0));
        v16bf a;
#pragma unroll
        for (int i = 0; i < 8; ++i) { a[i] = alo[i]; a[8 + i] = ahi[i]; }

        // cover HBM latency on the streamed A operand (global_prefetch_b8)
        if (kk + 256 < D_MODEL) __builtin_prefetch(aptr + kk + 256, 0, 0);

#pragma unroll
        for (int nt = 0; nt < 4; ++nt) {
            // B fragment (32x16): column n = W row (n0+nt*16+ln); lanes 0-15
            // hold K=kk..kk+15, lanes 16-31 hold K=kk+16..kk+31 (contiguous).
            const bf16* wptr =
                W + (size_t)(n0 + nt * 16 + ln) * D_MODEL + kk + (half ? 16 : 0);
            v16bf b = *(const v16bf*)wptr;
            acc[nt] = __builtin_amdgcn_wmma_f32_16x16x32_bf16(
                false, a, false, b, (short)0, acc[nt], false, false);
        }
    }

    // Epilogue. C layout: VGPR r holds row (r + half*8), column = n-tile + ln.
#pragma unroll
    for (int nt = 0; nt < 4; ++nt) {
#pragma unroll
        for (int r = 0; r < 8; ++r) {
            const int m = m0 + r + half * 8;
            const int n = n0 + nt * 16 + ln;
            const float v = (acc[nt][r] + bias[n]) * scale;
            if (MODE == 0) {
                const int b = m >> 11, s = m & (SEQ - 1);
                const int h = n >> 6,  d = n & (DH - 1);
                ((bf16*)out)[(((size_t)(b * NHEAD + h)) * SEQ + s) * DH + d] = (bf16)v;
            } else if (MODE == 1) {
                const int b = m >> 11, s = m & (SEQ - 1);
                const int h = n >> 6,  d = n & (DH - 1);
                ((bf16*)out)[(((size_t)(b * NHEAD + h)) * DH + d) * SEQ + s] = (bf16)v;
            } else {
                ((float*)out)[(size_t)m * D_MODEL + n] = v;
            }
        }
    }
}

// ---------------------------------------------------------------------------
// Flash attention, fully transposed formulation. One wave per 16-query tile,
// 4 independent waves per block (no barriers, no LDS in the hot loop).
//
//   Lt = K @ Q^T  : C layout => q along lanes, keys along VGPRs
//                   -> softmax stats are lane-local + one shfl_xor(16)
//   Ot = V^T @ P^T: rescale factor and 1/l are per-lane scalars
//   P^T B-fragment is built from logit registers with 8 packed shfl_xor(16).
//
// Q is pre-scaled by (1/sqrt(Dh))*log2(e) so softmax runs in base-2 domain
// with a bare subtract + raw v_exp_f32 per element.
//
//   q,k: (B,H,S,Dh) bf16;  vt: (B,H,Dh,S) bf16;  attn: (B,S,D) bf16
// ---------------------------------------------------------------------------
__global__ void __launch_bounds__(128)
flash_attn_wmma(const bf16* __restrict__ q, const bf16* __restrict__ k,
                const bf16* __restrict__ vt, bf16* __restrict__ attn) {
    const int lane = threadIdx.x & 31;
    const int wave = threadIdx.x >> 5;
    const int half = lane >> 4;
    const int ln   = lane & 15;
    const int q0   = (blockIdx.x * 4 + wave) * 16;
    const int h    = blockIdx.y;
    const int b    = blockIdx.z;

    const bf16* qh  = q  + ((size_t)(b * NHEAD + h)) * SEQ * DH;
    const bf16* kh  = k  + ((size_t)(b * NHEAD + h)) * SEQ * DH;
    const bf16* vth = vt + ((size_t)(b * NHEAD + h)) * DH * SEQ;

    // Q^T B-fragments (loop invariant): B[k=d][n=q], lane holds column q=ln,
    // K = d, contiguous 16 elements from the Q row.
    const bf16* qrow = qh + (size_t)(q0 + ln) * DH;
    v16bf bq0 = *(const v16bf*)(qrow + (half ? 16 : 0));
    v16bf bq1 = *(const v16bf*)(qrow + 32 + (half ? 16 : 0));

    v8f acc[4] = {};                 // Ot tiles: d = nt*16 + r + half*8, q = ln
    float m_run = -INFINITY;         // per-lane running max (log2 domain), q = ln
    float l_run = 0.0f;              // per-lane partial softmax denominator

    for (int kb = 0; kb < SEQ; kb += 32) {
        // ---- Lt = K @ Q^T : two 16(key)x16(q) C tiles, K-dim = Dh = 64 ----
        v8f ct0 = {}, ct1 = {};
        {
            const bf16* kr = kh + (size_t)(kb + ln) * DH;   // A: keys kb..kb+15
            v8bf lo = *(const v8bf*)(kr + (half ? 16 : 0));
            v8bf hi = *(const v8bf*)(kr + 8 + (half ? 16 : 0));
            v16bf a;
#pragma unroll
            for (int i = 0; i < 8; ++i) { a[i] = lo[i]; a[8 + i] = hi[i]; }
            ct0 = __builtin_amdgcn_wmma_f32_16x16x32_bf16(false, a, false, bq0,
                                                          (short)0, ct0, false, false);
            lo = *(const v8bf*)(kr + 32 + (half ? 16 : 0));
            hi = *(const v8bf*)(kr + 40 + (half ? 16 : 0));
#pragma unroll
            for (int i = 0; i < 8; ++i) { a[i] = lo[i]; a[8 + i] = hi[i]; }
            ct0 = __builtin_amdgcn_wmma_f32_16x16x32_bf16(false, a, false, bq1,
                                                          (short)0, ct0, false, false);
        }
        {
            const bf16* kr = kh + (size_t)(kb + 16 + ln) * DH; // keys kb+16..kb+31
            v8bf lo = *(const v8bf*)(kr + (half ? 16 : 0));
            v8bf hi = *(const v8bf*)(kr + 8 + (half ? 16 : 0));
            v16bf a;
#pragma unroll
            for (int i = 0; i < 8; ++i) { a[i] = lo[i]; a[8 + i] = hi[i]; }
            ct1 = __builtin_amdgcn_wmma_f32_16x16x32_bf16(false, a, false, bq0,
                                                          (short)0, ct1, false, false);
            lo = *(const v8bf*)(kr + 32 + (half ? 16 : 0));
            hi = *(const v8bf*)(kr + 40 + (half ? 16 : 0));
#pragma unroll
            for (int i = 0; i < 8; ++i) { a[i] = lo[i]; a[8 + i] = hi[i]; }
            ct1 = __builtin_amdgcn_wmma_f32_16x16x32_bf16(false, a, false, bq1,
                                                          (short)0, ct1, false, false);
        }

        // prefetch next key/value block while this one is processed
        if (kb + 32 < SEQ) {
            __builtin_prefetch(kh + (size_t)(kb + 32 + ln) * DH, 0, 0);
            __builtin_prefetch(vth + (size_t)ln * SEQ + kb + 32, 0, 0);
        }

        // ---- online softmax, base-2 domain; q = ln is fixed per lane ----
        // Lane holds 16 of the 32 keys for its q; partner lane (xor 16) holds
        // the other 16. One cross-lane op completes the block max.
        float bm = fmaxf(ct0[0], ct0[1]);
#pragma unroll
        for (int r = 2; r < 8; ++r) bm = fmaxf(bm, ct0[r]);
#pragma unroll
        for (int r = 0; r < 8; ++r) bm = fmaxf(bm, ct1[r]);
        bm = fmaxf(bm, __shfl_xor(bm, 16));
        const float m_new = fmaxf(m_run, bm);
        const float corr  = __builtin_amdgcn_exp2f(m_run - m_new); // raw v_exp_f32
        m_run = m_new;

        float e0[8], e1[8];
        float rs = 0.0f;
#pragma unroll
        for (int r = 0; r < 8; ++r) {
            e0[r] = __builtin_amdgcn_exp2f(ct0[r] - m_new);
            e1[r] = __builtin_amdgcn_exp2f(ct1[r] - m_new);
            rs += e0[r] + e1[r];
        }
        l_run = l_run * corr + rs;   // cross-half combine deferred to the end

        // rescale running output (per-lane scalar factor)
#pragma unroll
        for (int nt = 0; nt < 4; ++nt)
#pragma unroll
            for (int r = 0; r < 8; ++r) acc[nt][r] *= corr;

        // ---- build P^T B-fragment (32key x 16q) from logit registers ----
        // lanes 0-15 own keys {0-7,16-23}; B layout needs {0-15} / {16-31}:
        // swap 8-key halves with partner lane via packed shfl_xor(16).
        int pk0[4], pk1[4];
#pragma unroll
        for (int j = 0; j < 4; ++j) {
            bf16x2 t0, t1;
            t0[0] = (bf16)e0[2 * j]; t0[1] = (bf16)e0[2 * j + 1];
            t1[0] = (bf16)e1[2 * j]; t1[1] = (bf16)e1[2 * j + 1];
            pk0[j] = __builtin_bit_cast(int, t0);
            pk1[j] = __builtin_bit_cast(int, t1);
        }
        v8i bi;
#pragma unroll
        for (int j = 0; j < 4; ++j) {
            const int x0 = __shfl_xor(pk0[j], 16);
            const int x1 = __shfl_xor(pk1[j], 16);
            bi[j]     = half ? x1 : pk0[j];   // K = 0-7   | 16-23
            bi[4 + j] = half ? pk1[j] : x0;   // K = 8-15  | 24-31
        }
        const v16bf bp = __builtin_bit_cast(v16bf, bi);

        // ---- Ot += V^T @ P^T  (V^T rows are contiguous in vt) ----
#pragma unroll
        for (int nt = 0; nt < 4; ++nt) {
            const bf16* vr = vth + (size_t)(nt * 16 + ln) * SEQ + kb;
            v8bf lo = *(const v8bf*)(vr + (half ? 16 : 0));
            v8bf hi = *(const v8bf*)(vr + 8 + (half ? 16 : 0));
            v16bf av;
#pragma unroll
            for (int i = 0; i < 8; ++i) { av[i] = lo[i]; av[8 + i] = hi[i]; }
            acc[nt] = __builtin_amdgcn_wmma_f32_16x16x32_bf16(
                false, av, false, bp, (short)0, acc[nt], false, false);
        }
    }

    // ---- finalize: combine halves of l, normalize, write (B,S,D) bf16 ----
    const float l_tot = l_run + __shfl_xor(l_run, 16);
    const float inv_l = __builtin_amdgcn_rcpf(l_tot);

    bf16* orow = attn + ((size_t)b * SEQ + q0 + ln) * D_MODEL + h * DH + half * 8;
#pragma unroll
    for (int nt = 0; nt < 4; ++nt) {
        v8bf o;
#pragma unroll
        for (int r = 0; r < 8; ++r) o[r] = (bf16)(acc[nt][r] * inv_l);
        *(v8bf*)(orow + nt * 16) = o;   // 8 contiguous d per lane
    }
}

// ---------------------------------------------------------------------------
extern "C" void kernel_launch(void* const* d_in, const int* in_sizes, int n_in,
                              void* d_out, int out_size, void* d_ws, size_t ws_size,
                              hipStream_t stream) {
    (void)in_sizes; (void)n_in; (void)out_size; (void)ws_size;

    const float* x  = (const float*)d_in[0];
    const float* Wq = (const float*)d_in[1];
    const float* bq = (const float*)d_in[2];
    const float* Wk = (const float*)d_in[3];
    const float* bk = (const float*)d_in[4];
    const float* Wv = (const float*)d_in[5];
    const float* bv = (const float*)d_in[6];
    const float* Wo = (const float*)d_in[7];
    const float* bo = (const float*)d_in[8];

    char* ws = (char*)d_ws;
    size_t off = 0;
    bf16* xb  = (bf16*)(ws + off); off += (size_t)MTOT * D_MODEL * 2;
    bf16* wqb = (bf16*)(ws + off); off += (size_t)D_MODEL * D_MODEL * 2;
    bf16* wkb = (bf16*)(ws + off); off += (size_t)D_MODEL * D_MODEL * 2;
    bf16* wvb = (bf16*)(ws + off); off += (size_t)D_MODEL * D_MODEL * 2;
    bf16* wob = (bf16*)(ws + off); off += (size_t)D_MODEL * D_MODEL * 2;
    bf16* qh  = (bf16*)(ws + off); off += (size_t)MTOT * D_MODEL * 2;
    bf16* kh  = (bf16*)(ws + off); off += (size_t)MTOT * D_MODEL * 2;
    bf16* vth = (bf16*)(ws + off); off += (size_t)MTOT * D_MODEL * 2;
    bf16* att = (bf16*)(ws + off); off += (size_t)MTOT * D_MODEL * 2;

    // f32 -> bf16 conversions
    cvt_f32_bf16<<<1024, 256, 0, stream>>>(x, xb, MTOT * D_MODEL);
    cvt_f32_bf16<<<256, 256, 0, stream>>>(Wq, wqb, D_MODEL * D_MODEL);
    cvt_f32_bf16<<<256, 256, 0, stream>>>(Wk, wkb, D_MODEL * D_MODEL);
    cvt_f32_bf16<<<256, 256, 0, stream>>>(Wv, wvb, D_MODEL * D_MODEL);
    cvt_f32_bf16<<<256, 256, 0, stream>>>(Wo, wob, D_MODEL * D_MODEL);

    const float ls = 0.18033688f;    // (1/sqrt(64)) * log2(e), folded into Q

    // QKV projections (V stored transposed per head)
    dim3 gg(MTOT / 128, D_MODEL / 64);
    gemm_bias_wmma<0><<<gg, 256, 0, stream>>>(xb, wqb, bq, qh, ls);
    gemm_bias_wmma<0><<<gg, 256, 0, stream>>>(xb, wkb, bk, kh, 1.0f);
    gemm_bias_wmma<1><<<gg, 256, 0, stream>>>(xb, wvb, bv, vth, 1.0f);

    // attention: 4 waves/block, one 16-query tile per wave
    dim3 ga(SEQ / 64, NHEAD, BATCH);
    flash_attn_wmma<<<ga, 128, 0, stream>>>(qh, kh, vth, att);

    // output projection -> f32
    gemm_bias_wmma<2><<<gg, 256, 0, stream>>>(att, wob, bo, d_out, 1.0f);
}
